// cross_attention_lstm_multiple_layers_83141976916061
// MI455X (gfx1250) — compile-verified, matrix operand order checked by reading, not run on hardware
//
#include <hip/hip_runtime.h>
#include <hip/hip_bf16.h>
#include <math.h>

#define Bsz 32
#define Hd  512
#define Ld  2
#define Tm  20
#define Vv  10000
#define Vpad 10240   // 640 tiles of 16 -> divisible by NB=4 groups
#define Ss  49

typedef __attribute__((ext_vector_type(16))) __bf16 v16bf;
typedef __attribute__((ext_vector_type(8)))  __bf16 v8bf;
typedef __attribute__((ext_vector_type(8)))  float  v8f;

__device__ __forceinline__ float sigf(float x){ return 1.0f/(1.0f+__expf(-x)); }

// ---------------- fp32 -> bf16 convert, zero-padding past n_src ----------------
__global__ void cvt_pad_bf16(const float* __restrict__ s, __bf16* __restrict__ d,
                             int n_src, int n_total){
  for (int i = blockIdx.x*blockDim.x + threadIdx.x; i < n_total; i += gridDim.x*blockDim.x)
    d[i] = (i < n_src) ? (__bf16)s[i] : (__bf16)0.0f;
}

// ---------------- initial state ----------------
__global__ void init_state(const float* __restrict__ pooled, const float* __restrict__ embed,
                           const int* __restrict__ sos, float* __restrict__ c,
                           __bf16* __restrict__ h_bf, __bf16* __restrict__ pred0,
                           __bf16* __restrict__ emb_bf){
  int i = blockIdx.x*blockDim.x + threadIdx.x;
  if (i < Ld*Bsz*Hd){
    int b = (i/Hd)%Bsz, j = i%Hd;
    float p = pooled[b*Hd+j];
    c[i] = p; h_bf[i] = (__bf16)p;
  }
  if (i < Bsz*Hd){
    int j = i%Hd;
    float e = embed[(size_t)sos[0]*Hd + j];
    pred0[i] = (__bf16)e; emb_bf[i] = (__bf16)e;
  }
}

// ---------------- keys/values: tanh(chan @ Wk^T + bk), step-invariant ----------------
__global__ void kv_kernel(const float* __restrict__ chan, const float* __restrict__ Wk,
                          const float* __restrict__ bk, const float* __restrict__ Wv,
                          const float* __restrict__ bv, float* __restrict__ keys,
                          float* __restrict__ values){
  int i = blockIdx.x*blockDim.x + threadIdx.x;
  if (i >= Bsz*Hd*Ss) return;
  int s = i % Ss; int bd = i / Ss;
  const float* row = chan + (size_t)bd*Ss;
  float ak = bk[s], av = bv[s];
  for (int sp = 0; sp < Ss; ++sp){
    float x = row[sp];
    ak += x * Wk[s*Ss+sp];
    av += x * Wv[s*Ss+sp];
  }
  keys[i] = tanhf(ak); values[i] = tanhf(av);
}

// ---------------- WMMA bf16 GEMM: C[M,N] = A[M,K] @ W[N,K]^T (+A2@W2^T) + biases ----------------
// ISA 16-bit A-fragment layout: lane = half*16 + r holds row r,
// element e (0..15): K = kb + e%8 + (e/8)*16 + half*8  -> two 16B contiguous loads.
__device__ __forceinline__ v16bf load_frag(const __bf16* __restrict__ base, int ld,
                                           int row, int kb, int half){
  const __bf16* p = base + (size_t)row*ld + kb + half*8;
  v8bf lo = *(const v8bf*)p;
  v8bf hi = *(const v8bf*)(p + 16);
  v16bf r;
  #pragma unroll
  for (int e = 0; e < 8; ++e){ r[e] = lo[e]; r[8+e] = hi[e]; }
  return r;
}

// NB = N-tiles per wave (register blocking; A fragment reused NB times per k-chunk).
// REQUIREMENT: (N/16) % NB == 0 (pad W with zero rows); Nstore <= N limits the
// epilogue stores (tile-aligned boundary -> wave-uniform guard, no divergence).
template<int NB>
__global__ void gemm_wmma_bf16(
    const __bf16* __restrict__ A1, int lda1, const __bf16* __restrict__ W1,
    const __bf16* __restrict__ A2, int lda2, const __bf16* __restrict__ W2,
    const float* __restrict__ bias1, const float* __restrict__ bias2,
    float* __restrict__ Cf, int ldc, __bf16* __restrict__ Cbf, int ldcb,
    int M, int N, int K, int Nstore, int act, int out_mode)
{
  int wid  = (blockIdx.x * blockDim.x + threadIdx.x) >> 5;
  int lane = threadIdx.x & 31;
  int tilesN = N >> 4, tilesM = M >> 4;
  int nGroups = tilesN / NB;
  if (wid >= tilesM*nGroups) return;
  int tm = wid / nGroups, tn0 = (wid % nGroups)*NB;
  int r = lane & 15, half = lane >> 4;
  int arow = tm*16 + r;

  v8f acc[NB];
  #pragma unroll
  for (int i = 0; i < NB; ++i) acc[i] = (v8f){};

  for (int kb = 0; kb < K; kb += 32){
    v16bf a = load_frag(A1, lda1, arow, kb, half);
    #pragma unroll
    for (int i = 0; i < NB; ++i){
      v16bf w = load_frag(W1, K, (tn0+i)*16 + r, kb, half);
      acc[i] = __builtin_amdgcn_wmma_f32_16x16x32_bf16(false, a, false, w, (short)0, acc[i], false, false);
    }
  }
  if (A2){
    for (int kb = 0; kb < K; kb += 32){
      v16bf a = load_frag(A2, lda2, arow, kb, half);
      #pragma unroll
      for (int i = 0; i < NB; ++i){
        v16bf w = load_frag(W2, K, (tn0+i)*16 + r, kb, half);
        acc[i] = __builtin_amdgcn_wmma_f32_16x16x32_bf16(false, a, false, w, (short)0, acc[i], false, false);
      }
    }
  }

  #pragma unroll
  for (int i = 0; i < NB; ++i){
    int col = (tn0+i)*16 + r;
    if (col >= Nstore) continue;          // tile-aligned boundary: uniform per tile
    float badd = 0.f;
    if (bias1) badd += bias1[col];
    if (bias2) badd += bias2[col];
    #pragma unroll
    for (int v = 0; v < 8; ++v){
      int rowm = tm*16 + v + 8*half;      // C/D layout: M = v + 8*half
      float val = acc[i][v] + badd;
      if (act == 1) val = tanhf(val);
      if (out_mode == 1){                 // res[b, v, t], rowm = t*B + b
        int t = rowm / Bsz, b = rowm % Bsz;
        Cf[(size_t)b*((size_t)Vv*Tm) + (size_t)col*Tm + t] = val;
      } else {
        if (Cf)  Cf[(size_t)rowm*ldc + col] = val;
        if (Cbf) Cbf[(size_t)rowm*ldcb + col] = (__bf16)val;
      }
    }
  }
}

// ---------------- LSTM elementwise update ----------------
__global__ void lstm_update(const float* __restrict__ g, float* __restrict__ c,
                            __bf16* __restrict__ cat, __bf16* __restrict__ predrow, int l){
  int i = blockIdx.x*blockDim.x + threadIdx.x;
  if (i >= Bsz*Hd) return;
  int b = i / Hd, j = i % Hd;
  const float* gr = g + (size_t)b*4*Hd;
  float gi = gr[j], gf = gr[Hd+j], gg = gr[2*Hd+j], go = gr[3*Hd+j];
  size_t ci = (size_t)l*Bsz*Hd + i;
  float c2 = sigf(gf)*c[ci] + sigf(gi)*tanhf(gg);
  float h  = sigf(go)*tanhf(c2);
  c[ci] = c2;
  cat[(size_t)(l*Bsz+b)*(2*Hd) + Hd + j] = (__bf16)h;   // h half of concat (also q input / next-layer x)
  if (predrow) predrow[i] = (__bf16)h;                  // top-layer "out"
}

// ---------------- greedy argmax over V + embedding gather ----------------
__global__ void argmax_gather(const float* __restrict__ logits,
                              const __bf16* __restrict__ embed_bf,
                              __bf16* __restrict__ emb_next){
  int b = blockIdx.x;
  __shared__ float sv[256]; __shared__ int si[256];
  float best = -3.402823466e38f; int bi = 0;
  for (int v = threadIdx.x; v < Vv; v += 256){
    float x = logits[(size_t)b*Vv + v];
    if (x > best){ best = x; bi = v; }
  }
  sv[threadIdx.x] = best; si[threadIdx.x] = bi;
  __syncthreads();
  for (int off = 128; off > 0; off >>= 1){
    if (threadIdx.x < off){
      float o = sv[threadIdx.x+off]; int oi = si[threadIdx.x+off];
      if (o > sv[threadIdx.x] || (o == sv[threadIdx.x] && oi < si[threadIdx.x])){
        sv[threadIdx.x] = o; si[threadIdx.x] = oi;
      }
    }
    __syncthreads();
  }
  int idx = si[0];
  for (int j = threadIdx.x; j < Hd; j += 256)
    emb_next[(size_t)b*Hd + j] = embed_bf[(size_t)idx*Hd + j];
}

// ---------------- 49-wide cross-attention (scores + softmax + weighted sum) ----------------
__global__ void attention_kernel(const float* __restrict__ q, const float* __restrict__ keys,
                                 const float* __restrict__ values, __bf16* __restrict__ cat){
  int lb = blockIdx.x;            // 0..L*B-1
  int b  = lb % Bsz;
  __shared__ float w[Ss];
  __shared__ float inv_s;
  int t = threadIdx.x;            // 64 threads
  if (t < Ss){
    const float* qr = q + (size_t)lb*Hd;
    const float* kp = keys + (size_t)b*Hd*Ss + t;
    float s = 0.f;
    for (int d = 0; d < Hd; ++d) s += qr[d]*kp[(size_t)d*Ss];
    w[t] = s * (1.0f/7.0f);
  }
  __syncthreads();
  if (t == 0){
    float m = w[0];
    for (int s2 = 1; s2 < Ss; ++s2) m = fmaxf(m, w[s2]);
    float sum = 0.f;
    for (int s2 = 0; s2 < Ss; ++s2){ w[s2] = __expf(w[s2]-m); sum += w[s2]; }
    inv_s = 1.0f/sum;
  }
  __syncthreads();
  float inv = inv_s;
  const float* vb = values + (size_t)b*Hd*Ss;
  for (int d = t; d < Hd; d += 64){
    float a = 0.f;
    for (int s2 = 0; s2 < Ss; ++s2) a += w[s2]*vb[(size_t)d*Ss+s2];
    cat[(size_t)lb*(2*Hd) + d] = (__bf16)(a*inv);       // attn half of concat
  }
}

// ---------------- host ----------------
static void launch_gemm(const __bf16* A1, int lda1, const __bf16* W1,
                        const __bf16* A2, int lda2, const __bf16* W2,
                        const float* b1, const float* b2,
                        float* Cf, int ldc, __bf16* Cbf, int ldcb,
                        int M, int Npad, int K, int Nstore, int act, int out_mode,
                        hipStream_t stream){
  int tilesM = M/16, tilesN = Npad/16;
  if (tilesN % 4 == 0 && Npad >= 4096){
    int waves = tilesM * (tilesN/4);
    int blocks = (waves*32 + 255)/256;
    gemm_wmma_bf16<4><<<blocks, 256, 0, stream>>>(A1, lda1, W1, A2, lda2, W2,
                                                  b1, b2, Cf, ldc, Cbf, ldcb,
                                                  M, Npad, K, Nstore, act, out_mode);
  } else {
    int waves = tilesM * tilesN;
    int blocks = (waves*32 + 255)/256;
    gemm_wmma_bf16<1><<<blocks, 256, 0, stream>>>(A1, lda1, W1, A2, lda2, W2,
                                                  b1, b2, Cf, ldc, Cbf, ldcb,
                                                  M, Npad, K, Nstore, act, out_mode);
  }
}

extern "C" void kernel_launch(void* const* d_in, const int* in_sizes, int n_in,
                              void* d_out, int out_size, void* d_ws, size_t ws_size,
                              hipStream_t stream) {
  const float* img    = (const float*)d_in[0];
  const float* pooled = (const float*)d_in[1];
  const float* embed  = (const float*)d_in[2];
  const float* Wq     = (const float*)d_in[3];
  const float* bq     = (const float*)d_in[4];
  const float* Wk     = (const float*)d_in[5];
  const float* bk     = (const float*)d_in[6];
  const float* Wv     = (const float*)d_in[7];
  const float* bv     = (const float*)d_in[8];
  const float* Wih    = (const float*)d_in[9];
  const float* Whh    = (const float*)d_in[10];
  const float* bih    = (const float*)d_in[11];
  const float* bhh    = (const float*)d_in[12];
  const float* projW  = (const float*)d_in[13];
  const float* projb  = (const float*)d_in[14];
  const float* hattW  = (const float*)d_in[15];
  const float* hattb  = (const float*)d_in[16];
  const int*   sos    = (const int*)d_in[17];
  float* out = (float*)d_out;

  char* p = (char*)d_ws;
  auto alloc = [&](size_t bytes)->char*{
    char* r = p; p += (bytes + 255) & ~(size_t)255; return r;
  };
  __bf16* wih_bf   = (__bf16*)alloc(sizeof(__bf16)*(size_t)Ld*4*Hd*Hd);
  __bf16* whh_bf   = (__bf16*)alloc(sizeof(__bf16)*(size_t)Ld*4*Hd*Hd);
  __bf16* proj_bf  = (__bf16*)alloc(sizeof(__bf16)*(size_t)Vpad*Hd);  // zero-padded rows
  __bf16* wq_bf    = (__bf16*)alloc(sizeof(__bf16)*(size_t)Hd*Hd);
  __bf16* hatt_bf  = (__bf16*)alloc(sizeof(__bf16)*(size_t)Hd*2*Hd);
  __bf16* embed_bf = (__bf16*)alloc(sizeof(__bf16)*(size_t)Vv*Hd);
  float*  keys     = (float*) alloc(sizeof(float)*(size_t)Bsz*Hd*Ss);
  float*  values   = (float*) alloc(sizeof(float)*(size_t)Bsz*Hd*Ss);
  float*  cbuf     = (float*) alloc(sizeof(float)*(size_t)Ld*Bsz*Hd);
  __bf16* h_bf     = (__bf16*)alloc(sizeof(__bf16)*(size_t)Ld*Bsz*Hd);
  __bf16* cat      = (__bf16*)alloc(sizeof(__bf16)*(size_t)Ld*Bsz*2*Hd);
  float*  qbuf     = (float*) alloc(sizeof(float)*(size_t)Ld*Bsz*Hd);
  float*  g        = (float*) alloc(sizeof(float)*(size_t)Bsz*4*Hd);
  float*  logits   = (float*) alloc(sizeof(float)*(size_t)Bsz*Vv);
  __bf16* emb_bf   = (__bf16*)alloc(sizeof(__bf16)*(size_t)Bsz*Hd);
  __bf16* pred_bf  = (__bf16*)alloc(sizeof(__bf16)*(size_t)Tm*Bsz*Hd);

  auto cvt = [&](const float* s, __bf16* d, int n_src, int n_total){
    int blocks = (n_total + 255)/256; if (blocks > 4096) blocks = 4096;
    cvt_pad_bf16<<<blocks, 256, 0, stream>>>(s, d, n_src, n_total);
  };
  cvt(Wih,   wih_bf,   Ld*4*Hd*Hd, Ld*4*Hd*Hd);
  cvt(Whh,   whh_bf,   Ld*4*Hd*Hd, Ld*4*Hd*Hd);
  cvt(projW, proj_bf,  Vv*Hd,      Vpad*Hd);   // zero rows 10000..10239
  cvt(Wq,    wq_bf,    Hd*Hd,      Hd*Hd);
  cvt(hattW, hatt_bf,  Hd*2*Hd,    Hd*2*Hd);
  cvt(embed, embed_bf, Vv*Hd,      Vv*Hd);

  init_state<<<(Ld*Bsz*Hd + 255)/256, 256, 0, stream>>>(pooled, embed, sos,
                                                        cbuf, h_bf, pred_bf, emb_bf);
  kv_kernel<<<(Bsz*Hd*Ss + 255)/256, 256, 0, stream>>>(img, Wk, bk, Wv, bv, keys, values);

  const int eu = (Bsz*Hd + 255)/256;
  for (int t = 0; t < Tm-1; ++t){
    // layer 0 gates: emb @ Wih0^T + h0 @ Whh0^T
    launch_gemm(emb_bf, Hd, wih_bf, h_bf, Hd, whh_bf,
                bih, bhh, g, 4*Hd, nullptr, 0, Bsz, 4*Hd, Hd, 4*Hd, 0, 0, stream);
    lstm_update<<<eu, 256, 0, stream>>>(g, cbuf, cat, nullptr, 0);
    // layer 1 gates: h0_new (cat rows 0..31, cols 512..) + h1
    launch_gemm(cat + Hd, 2*Hd, wih_bf + (size_t)4*Hd*Hd,
                h_bf + (size_t)Bsz*Hd, Hd, whh_bf + (size_t)4*Hd*Hd,
                bih + 4*Hd, bhh + 4*Hd, g, 4*Hd, nullptr, 0, Bsz, 4*Hd, Hd, 4*Hd, 0, 0, stream);
    lstm_update<<<eu, 256, 0, stream>>>(g, cbuf, cat,
                                        pred_bf + (size_t)(t+1)*Bsz*Hd, 1);
    // greedy decode: logits = out @ projW^T + projb  (NB=4 over padded 10240)
    launch_gemm(pred_bf + (size_t)(t+1)*Bsz*Hd, Hd, proj_bf, nullptr, 0, nullptr,
                projb, nullptr, logits, Vv, nullptr, 0, Bsz, Vpad, Hd, Vv, 0, 0, stream);
    argmax_gather<<<Bsz, 256, 0, stream>>>(logits, embed_bf, emb_bf);
    // q = tanh(h @ Wq^T + bq), both layers at once (rows = l*B+b)
    launch_gemm(cat + Hd, 2*Hd, wq_bf, nullptr, 0, nullptr,
                bq, nullptr, qbuf, Hd, nullptr, 0, Ld*Bsz, Hd, Hd, Hd, 1, 0, stream);
    attention_kernel<<<Ld*Bsz, 64, 0, stream>>>(qbuf, keys, values, cat);
    // h = tanh([attn, h] @ hattW^T + hattb) -> bf16 h for next step
    launch_gemm(cat, 2*Hd, hatt_bf, nullptr, 0, nullptr,
                hattb, nullptr, nullptr, 0, h_bf, Hd, Ld*Bsz, Hd, 2*Hd, Hd, 1, 0, stream);
  }

  // res[b,v,t] = pred[t,b,:] @ projW^T + projb  (NB=4 over padded 10240, scatter epilogue)
  launch_gemm(pred_bf, Hd, proj_bf, nullptr, 0, nullptr,
              projb, nullptr, out, 0, nullptr, 0, Tm*Bsz, Vpad, Hd, Vv, 0, 1, stream);
}
